// EuclideanHead_9397388443804
// MI455X (gfx1250) — matrix operand order repaired and checked by reading, not compiled, not run on hardware
//
#include <hip/hip_runtime.h>
#include <hip/hip_bf16.h>

typedef __attribute__((ext_vector_type(16))) _Float16 v16h;
typedef __attribute__((ext_vector_type(8)))  _Float16 v8h;
typedef __attribute__((ext_vector_type(4)))  _Float16 v4h;
typedef __attribute__((ext_vector_type(8)))  float    v8f;
typedef __attribute__((ext_vector_type(4)))  int      v4i;

// Pointer types matching the async-copy builtin's signature:
// (int4 AS(1)*, int4 AS(3)*, imm offset, imm cpol)
typedef __attribute__((address_space(1))) v4i* gv4i_p;  // global int4*
typedef __attribute__((address_space(3))) v4i* lv4i_p;  // LDS int4*

#if defined(__HIP_DEVICE_COMPILE__) && defined(__gfx1250__) &&            \
    __has_builtin(__builtin_amdgcn_global_load_async_to_lds_b128) &&      \
    __has_builtin(__builtin_amdgcn_s_wait_asynccnt)
#define USE_ASYNC_LDS 1
#else
#define USE_ASYNC_LDS 0
#endif

namespace {
constexpr int kN  = 8192;
constexpr int kD  = 512;
constexpr int kBM = 128;            // tile rows per block
constexpr int kBN = 128;            // tile cols per block
constexpr int kKC = 32;             // K chunk = one WMMA K step
constexpr int kP  = kKC + 8;        // LDS pitch (halves): 80B row stride, de-banks gathers
constexpr int kIters = kD / kKC;    // 16
}

union FragU { v16h v; v8h h[2]; };

// 16B global -> LDS copy: async ASYNCcnt-tracked path when available.
__device__ __forceinline__ void cp16(const _Float16* g, _Float16* l) {
#if USE_ASYNC_LDS
  __builtin_amdgcn_global_load_async_to_lds_b128(
      (gv4i_p)(g), (lv4i_p)(l), 0, 0);
#else
  *(v8h*)l = *(const v8h*)g;
#endif
}

__device__ __forceinline__ void wait_fills() {
#if USE_ASYNC_LDS
  __builtin_amdgcn_s_wait_asynccnt(0);
#endif
}

// A-operand fragment (16x32 f16). Lanes 0-15: row M=lane, K=0..7 & 16..23;
// lanes 16-31: same rows, K=8..15 & 24..31.  plane is [rows][kP] in LDS.
__device__ __forceinline__ v16h load_frag_A(const _Float16* plane, int rowbase, int lane) {
  const int row = rowbase + (lane & 15);
  const int kb  = (lane >> 4) << 3;
  const _Float16* p = plane + row * kP + kb;
  FragU u;
  u.h[0] = *(const v8h*)(p);
  u.h[1] = *(const v8h*)(p + 16);
  return u.v;
}

// B-operand fragment (32x16 f16). Lanes 0-15: col N=lane, K=0..7 & 16..23; lanes
// 16-31: N=lane-16, K=8..15 & 24..31. Gram matrix => B cols are again rows of E.
__device__ __forceinline__ v16h load_frag_B(const _Float16* plane, int rowbase, int lane) {
  const int row = rowbase + (lane & 15);
  const int kb  = (lane >> 4) << 3;
  const _Float16* p = plane + row * kP + kb;
  FragU u;
  u.h[0] = *(const v8h*)(p);
  u.h[1] = *(const v8h*)(p + 16);
  return u.v;
}

// ---------------------------------------------------------------------------
// Prepass: split fp32 E into f16 hi/lo planes and compute row squared norms.
// ---------------------------------------------------------------------------
__global__ __launch_bounds__(128) void euclid_prep_kernel(
    const float* __restrict__ E, _Float16* __restrict__ Ehi,
    _Float16* __restrict__ Elo, float* __restrict__ sq) {
  __shared__ float red[128];
  const int row = blockIdx.x;
  const int t   = threadIdx.x;

  const float4 x = ((const float4*)(E + (size_t)row * kD))[t];
  const float xs[4] = {x.x, x.y, x.z, x.w};
  v4h hh, ll;
  float s = 0.0f;
#pragma unroll
  for (int i = 0; i < 4; ++i) {
    const float v = xs[i];
    const _Float16 h = (_Float16)v;
    hh[i] = h;
    ll[i] = (_Float16)(v - (float)h);
    s += v * v;
  }
  ((v4h*)(Ehi + (size_t)row * kD))[t] = hh;
  ((v4h*)(Elo + (size_t)row * kD))[t] = ll;

  red[t] = s;
  __syncthreads();
#pragma unroll
  for (int off = 64; off > 0; off >>= 1) {
    if (t < off) red[t] += red[t + off];
    __syncthreads();
  }
  if (t == 0) sq[row] = red[0];
}

// ---------------------------------------------------------------------------
// Main kernel: 128x128 tile per block, 8 waves (2M x 4N), 64x32 per wave.
// Double-buffered LDS, async global->LDS fills overlapped with WMMA math.
// acc += Ahi*Bhi + Ahi*Blo + Alo*Bhi (f32 accumulate, split-f16 precision).
// ---------------------------------------------------------------------------
__global__ __launch_bounds__(256) void euclid_dist_kernel(
    const _Float16* __restrict__ Ehi, const _Float16* __restrict__ Elo,
    const float* __restrict__ sq, float* __restrict__ out) {
  __shared__ _Float16 sAhi[2][kBM * kP];
  __shared__ _Float16 sAlo[2][kBM * kP];
  __shared__ _Float16 sBhi[2][kBN * kP];
  __shared__ _Float16 sBlo[2][kBN * kP];
  __shared__ float sSqM[kBM];
  __shared__ float sSqN[kBN];

  const int tid  = threadIdx.x;
  const int lane = tid & 31;
  const int wave = tid >> 5;
  const int wm   = wave >> 2;                 // 0..1  -> 64-row slice
  const int wn   = wave & 3;                  // 0..3  -> 32-col slice
  const int row0 = blockIdx.y * kBM;
  const int col0 = blockIdx.x * kBN;

  if (tid < kBM) sSqM[tid]       = sq[row0 + tid];
  else           sSqN[tid - kBM] = sq[col0 + (tid - kBM)];

  v8f acc[4][2];
#pragma unroll
  for (int mf = 0; mf < 4; ++mf)
#pragma unroll
    for (int nf = 0; nf < 2; ++nf) acc[mf][nf] = (v8f){};

  // Fill mapping: 2 threads per row, 16 halves (2 x b128) per array each.
  const int tr = tid >> 1;
  const int tc = (tid & 1) << 4;              // 0 or 16 halves
  const size_t aRow = (size_t)(row0 + tr) * kD;
  const size_t bRow = (size_t)(col0 + tr) * kD;

  auto fill = [&](int buf, int kt) {
    const size_t go = (size_t)kt + tc;
    const _Float16* gAh = Ehi + aRow + go;
    const _Float16* gAl = Elo + aRow + go;
    const _Float16* gBh = Ehi + bRow + go;
    const _Float16* gBl = Elo + bRow + go;
    const int lo = tr * kP + tc;
    cp16(gAh,     &sAhi[buf][lo]);
    cp16(gAh + 8, &sAhi[buf][lo + 8]);
    cp16(gAl,     &sAlo[buf][lo]);
    cp16(gAl + 8, &sAlo[buf][lo + 8]);
    cp16(gBh,     &sBhi[buf][lo]);
    cp16(gBh + 8, &sBhi[buf][lo + 8]);
    cp16(gBl,     &sBlo[buf][lo]);
    cp16(gBl + 8, &sBlo[buf][lo + 8]);
  };

  fill(0, 0);                                  // prologue: chunk 0 in flight

  for (int c = 0; c < kIters; ++c) {
    wait_fills();                              // this wave's fill(c) landed
    __syncthreads();                           // all waves: fill(c) visible,
                                               // compute(c-1) fully done
    if (c + 1 < kIters) fill((c + 1) & 1, (c + 1) * kKC);  // overlaps math

    const int buf = c & 1;
    v16h ah[4], al[4];
#pragma unroll
    for (int mf = 0; mf < 4; ++mf) {
      ah[mf] = load_frag_A(&sAhi[buf][0], wm * 64 + mf * 16, lane);
      al[mf] = load_frag_A(&sAlo[buf][0], wm * 64 + mf * 16, lane);
    }
    v16h bh[2], bl[2];
#pragma unroll
    for (int nf = 0; nf < 2; ++nf) {
      bh[nf] = load_frag_B(&sBhi[buf][0], wn * 32 + nf * 16, lane);
      bl[nf] = load_frag_B(&sBlo[buf][0], wn * 32 + nf * 16, lane);
    }
#pragma unroll
    for (int mf = 0; mf < 4; ++mf)
#pragma unroll
      for (int nf = 0; nf < 2; ++nf) {
        acc[mf][nf] = __builtin_amdgcn_wmma_f32_16x16x32_f16(
            false, ah[mf], false, bh[nf], (short)0, acc[mf][nf], false, false);
        acc[mf][nf] = __builtin_amdgcn_wmma_f32_16x16x32_f16(
            false, ah[mf], false, bl[nf], (short)0, acc[mf][nf], false, false);
        acc[mf][nf] = __builtin_amdgcn_wmma_f32_16x16x32_f16(
            false, al[mf], false, bh[nf], (short)0, acc[mf][nf], false, false);
      }
  }

  // Epilogue: d = sqrt(max(||a||^2 + ||b||^2 - 2 a.b, 0)).  C/D layout:
  // VGPR r, lanes 0-15 -> (M=r, N=lane); lanes 16-31 -> (M=r+8, N=lane-16).
  const int lh = lane >> 4;
  const int nc = lane & 15;
#pragma unroll
  for (int mf = 0; mf < 4; ++mf)
#pragma unroll
    for (int nf = 0; nf < 2; ++nf)
#pragma unroll
      for (int r = 0; r < 8; ++r) {
        const int ml = wm * 64 + mf * 16 + lh * 8 + r;
        const int nl = wn * 32 + nf * 16 + nc;
        const float d2 = sSqM[ml] + sSqN[nl] - 2.0f * acc[mf][nf][r];
        const float dv = __builtin_sqrtf(__builtin_fmaxf(d2, 0.0f));
        // Write-once 268MB stream: keep it out of the L2 working set.
        __builtin_nontemporal_store(dv, &out[(size_t)(row0 + ml) * kN + (col0 + nl)]);
      }
}

extern "C" void kernel_launch(void* const* d_in, const int* in_sizes, int n_in,
                              void* d_out, int out_size, void* d_ws, size_t ws_size,
                              hipStream_t stream) {
  (void)in_sizes; (void)n_in; (void)out_size; (void)ws_size;
  const float* E   = (const float*)d_in[0];
  float*       out = (float*)d_out;

  // Workspace layout: Ehi (8 MB) | Elo (8 MB) | sq (32 KB)  -> ~16.03 MB.
  char* ws = (char*)d_ws;
  _Float16* Ehi = (_Float16*)(ws);
  _Float16* Elo = (_Float16*)(ws + (size_t)kN * kD * sizeof(_Float16));
  float*    sqv = (float*)(ws + (size_t)2 * kN * kD * sizeof(_Float16));

  euclid_prep_kernel<<<kN, 128, 0, stream>>>(E, Ehi, Elo, sqv);

  dim3 grid(kN / kBN, kN / kBM);   // 64 x 64 tiles
  euclid_dist_kernel<<<grid, 256, 0, stream>>>(Ehi, Elo, sqv, out);
}
